// GroupedQueryAttention_47450798686671
// MI455X (gfx1250) — matrix operand order repaired and checked by reading, not compiled, not run on hardware
//
#include <hip/hip_runtime.h>

// ---------------------------------------------------------------------------
// GQA forward for MI455X (gfx1250, wave32, WMMA).
// f32->f16 convert -> Q/K/V GEMMs -> flash attention (64-key blocks, LDS-staged
// K and packed-transposed V, base-2 online softmax, double-buffered fragments)
// -> output GEMM (+bias) in f32.
// ---------------------------------------------------------------------------

typedef __attribute__((ext_vector_type(16))) _Float16 v16h;
typedef __attribute__((ext_vector_type(8)))  float    v8f;

union Frag {
    uint4 u4[2];
    v16h  v;
};

__device__ inline unsigned short f32_to_f16_bits(float x) {
    _Float16 h = (_Float16)x;
    return __builtin_bit_cast(unsigned short, h);
}

__device__ inline v8f zero_v8f() {
    v8f z;
#pragma unroll
    for (int i = 0; i < 8; ++i) z[i] = 0.0f;
    return z;
}

// ---------------------------------------------------------------------------
// fp32 -> fp16 (bit pattern in ushort) elementwise
// ---------------------------------------------------------------------------
__global__ __launch_bounds__(256) void cvt_f32_to_f16(
    const float* __restrict__ in, unsigned short* __restrict__ out, int n)
{
    int i = blockIdx.x * 256 + threadIdx.x;
    if (i < n) out[i] = f32_to_f16_bits(in[i]);
}

// ---------------------------------------------------------------------------
// C[M,N] = A[M,K] @ W[K,N] (+bias). A,W f16 bits, f32 accum.
// Block: 128 threads (4 waves). Workgroup tile 128x64, wave tile 32x64,
// K-step 32. 8 WMMAs per wave per K-step.
// ---------------------------------------------------------------------------
template <bool OUT_F32>
__global__ __launch_bounds__(128) void gemm_f16_wmma(
    const unsigned short* __restrict__ A,
    const unsigned short* __restrict__ W,
    const float* __restrict__ bias,
    void* __restrict__ Cout, int M, int N, int K)
{
    __shared__ unsigned short As[128][40];  // [Mtile][K(32)+pad] 80B rows
    __shared__ unsigned short Bs[64][40];   // transposed: [Ntile][K(32)+pad]

    const int tid  = threadIdx.x;
    const int lane = tid & 31;
    const int wv   = tid >> 5;
    const int lo   = lane & 15;
    const int hi   = lane >> 4;
    const int m0   = blockIdx.y * 128;
    const int n0   = blockIdx.x * 64;

    v8f acc[2][4];
#pragma unroll
    for (int i = 0; i < 2; ++i)
#pragma unroll
        for (int s = 0; s < 4; ++s) acc[i][s] = zero_v8f();

    for (int k0 = 0; k0 < K; k0 += 32) {
        // Stage A tile 128x32 (coalesced uint4 rows)
#pragma unroll
        for (int p = 0; p < 4; ++p) {
            int idx = (p * 128 + tid) * 8;
            int row = idx >> 5, col = idx & 31;
            *(uint4*)&As[row][col] =
                *(const uint4*)&A[(size_t)(m0 + row) * K + k0 + col];
        }
        // Stage W tile 32x64 transposed into Bs[n][k]
#pragma unroll
        for (int p = 0; p < 2; ++p) {
            int idx = (p * 128 + tid) * 8;
            int kr = idx >> 6, nc = idx & 63;
            uint4 w4 = *(const uint4*)&W[(size_t)(k0 + kr) * N + n0 + nc];
            const unsigned short* tp = (const unsigned short*)&w4;
#pragma unroll
            for (int j = 0; j < 8; ++j) Bs[nc + j][kr] = tp[j];
        }
        __syncthreads();

        // Two A fragments (rows wv*32+lo and wv*32+16+lo)
        Frag a[2];
#pragma unroll
        for (int i = 0; i < 2; ++i) {
            a[i].u4[0] = *(const uint4*)&As[wv * 32 + i * 16 + lo][hi * 8];
            a[i].u4[1] = *(const uint4*)&As[wv * 32 + i * 16 + lo][16 + hi * 8];
        }
#pragma unroll
        for (int s = 0; s < 4; ++s) {
            Frag b;
            b.u4[0] = *(const uint4*)&Bs[s * 16 + lo][hi * 8];
            b.u4[1] = *(const uint4*)&Bs[s * 16 + lo][16 + hi * 8];
            acc[0][s] = __builtin_amdgcn_wmma_f32_16x16x32_f16(
                false, a[0].v, false, b.v, (short)0, acc[0][s], false, false);
            acc[1][s] = __builtin_amdgcn_wmma_f32_16x16x32_f16(
                false, a[1].v, false, b.v, (short)0, acc[1][s], false, false);
        }
        __syncthreads();
    }

#pragma unroll
    for (int i = 0; i < 2; ++i) {
#pragma unroll
        for (int s = 0; s < 4; ++s) {
            int col = n0 + s * 16 + lo;
            float bv = bias[col];
#pragma unroll
            for (int r = 0; r < 8; ++r) {
                int row = m0 + wv * 32 + i * 16 + hi * 8 + r;
                float val = acc[i][s][r] + bv;
                if (OUT_F32)
                    ((float*)Cout)[(size_t)row * N + col] = val;
                else
                    ((unsigned short*)Cout)[(size_t)row * N + col] =
                        f32_to_f16_bits(val);
            }
        }
    }
}

// ---------------------------------------------------------------------------
// Flash attention: block = (b, head, 64 query rows), 128 threads (4 waves),
// wave = 16 query rows. Key blocks of 64. K staged in LDS (shared by all
// waves), V staged transposed via packed b64 stores. Base-2 online softmax
// with scale*log2e folded into Q. 32 WMMAs per key block per wave.
// Q: [B*T, 2048] (head h at col h*128), K/V: [B*T, 512] (group g at g*128).
// ---------------------------------------------------------------------------
__global__ __launch_bounds__(128) void attn_wmma(
    const unsigned short* __restrict__ Kg,
    const unsigned short* __restrict__ Q,
    const unsigned short* __restrict__ V,
    unsigned short* __restrict__ O)
{
    __shared__ unsigned short Ks[64][144];    // K block natural [key][dim]
    __shared__ unsigned short Vt[128][88];    // V block transposed [dim][key]
    __shared__ unsigned short Pl[4][16][80];  // per-wave P tile [m][kloc]

    const int tid  = threadIdx.x;
    const int lane = tid & 31;
    const int wv   = tid >> 5;
    const int lo   = lane & 15;
    const int hi   = lane >> 4;

    const int q0 = blockIdx.x * 64;
    const int h  = blockIdx.y;
    const int b  = blockIdx.z;
    const int g  = h >> 2;                  // 4 heads per KV group
    const int T  = 2048;
    // 1/sqrt(128) * log2(e): folded into Q so scores are in log2 domain
    const _Float16 qscale = (_Float16)(0.08838834764831845f * 1.4426950408889634f);

    // Q fragments for this wave's 16 rows (hd=128 -> 4 K-steps), pre-scaled
    Frag qf[4];
    const unsigned short* qrow =
        Q + ((size_t)(b * T + q0 + wv * 16 + lo)) * 2048 + h * 128;
#pragma unroll
    for (int c = 0; c < 4; ++c) {
        qf[c].u4[0] = *(const uint4*)&qrow[c * 32 + hi * 8];
        qf[c].u4[1] = *(const uint4*)&qrow[c * 32 + 16 + hi * 8];
        qf[c].v     = qf[c].v * qscale;     // v_pk_mul_f16
    }

    float m_i[8], l_i[8];
    v8f oacc[8];
#pragma unroll
    for (int r = 0; r < 8; ++r) { m_i[r] = -1e30f; l_i[r] = 0.0f; }
#pragma unroll
    for (int c = 0; c < 8; ++c) oacc[c] = zero_v8f();

    // V staging coordinates: 4 consecutive keys x 8 dims per thread per pass
    const int vkg = tid & 15;        // key group (4 keys)
    const int vdc = (tid >> 4) * 8;  // dim chunk base (plus p*64)

    auto loadKfrag = [&](Frag& f, int nb, int c) {
        f.u4[0] = *(const uint4*)&Ks[nb * 16 + lo][c * 32 + hi * 8];
        f.u4[1] = *(const uint4*)&Ks[nb * 16 + lo][c * 32 + 16 + hi * 8];
    };
    auto loadVfrag = [&](Frag& f, int c, int hh) {
        f.u4[0] = *(const uint4*)&Vt[c * 16 + lo][hh * 32 + hi * 8];
        f.u4[1] = *(const uint4*)&Vt[c * 16 + lo][hh * 32 + 16 + hi * 8];
    };

    for (int kb = 0; kb < T; kb += 64) {
        // Prefetch next key block while this one computes
        if (kb + 64 < T) {
            __builtin_prefetch(
                &Kg[((size_t)(b * T + kb + 64 + (tid >> 1))) * 512 + g * 128 +
                    (tid & 1) * 64], 0, 3);
            __builtin_prefetch(
                &V[((size_t)(b * T + kb + 64 + (tid >> 1))) * 512 + g * 128 +
                   (tid & 1) * 64], 0, 3);
        }

        // Stage K block naturally: 64 keys x 128 dims, b128 in / b128 out
#pragma unroll
        for (int p = 0; p < 8; ++p) {
            int idx = (p * 128 + tid) * 8;
            int kr = idx >> 7, dc = idx & 127;
            uint4 k4 =
                *(const uint4*)&Kg[((size_t)(b * T + kb + kr)) * 512 + g * 128 + dc];
            *(uint4*)&Ks[kr][dc] = k4;
        }
        // Stage V transposed: load 4 key rows x 8 dims, pack 4 keys per b64
#pragma unroll
        for (int p = 0; p < 2; ++p) {
            int dc = vdc + p * 64;
            uint4 r0 = *(const uint4*)&V[((size_t)(b * T + kb + vkg * 4 + 0)) * 512 + g * 128 + dc];
            uint4 r1 = *(const uint4*)&V[((size_t)(b * T + kb + vkg * 4 + 1)) * 512 + g * 128 + dc];
            uint4 r2 = *(const uint4*)&V[((size_t)(b * T + kb + vkg * 4 + 2)) * 512 + g * 128 + dc];
            uint4 r3 = *(const uint4*)&V[((size_t)(b * T + kb + vkg * 4 + 3)) * 512 + g * 128 + dc];
            const unsigned short* a0 = (const unsigned short*)&r0;
            const unsigned short* a1 = (const unsigned short*)&r1;
            const unsigned short* a2 = (const unsigned short*)&r2;
            const unsigned short* a3 = (const unsigned short*)&r3;
#pragma unroll
            for (int d = 0; d < 8; ++d) {
                uint2 pk;
                pk.x = (unsigned)a0[d] | ((unsigned)a1[d] << 16);
                pk.y = (unsigned)a2[d] | ((unsigned)a3[d] << 16);
                *(uint2*)&Vt[dc + d][vkg * 4] = pk;
            }
        }
        __syncthreads();

        // Scores: four 16x16 key tiles, K-dim = 128 -> 16 WMMAs,
        // B fragments double-buffered from LDS.
        v8f sc[4];
#pragma unroll
        for (int nb = 0; nb < 4; ++nb) sc[nb] = zero_v8f();
        {
            Frag bfA, bfB;
            loadKfrag(bfA, 0, 0);
#pragma unroll
            for (int i = 0; i < 16; ++i) {
                int nb = i >> 2, c = i & 3;
                if ((i & 1) == 0) {
                    if (i + 1 < 16) loadKfrag(bfB, (i + 1) >> 2, (i + 1) & 3);
                    sc[nb] = __builtin_amdgcn_wmma_f32_16x16x32_f16(
                        false, qf[c].v, false, bfA.v, (short)0, sc[nb], false, false);
                } else {
                    if (i + 1 < 16) loadKfrag(bfA, (i + 1) >> 2, (i + 1) & 3);
                    sc[nb] = __builtin_amdgcn_wmma_f32_16x16x32_f16(
                        false, qf[c].v, false, bfB.v, (short)0, sc[nb], false, false);
                }
            }
        }

        // Online softmax (log2 domain); row m = hi*8+r lives in one 16-lane
        // half. P converted to f16 and stored to LDS inline.
#pragma unroll
        for (int r = 0; r < 8; ++r) {
            float mx = fmaxf(fmaxf(sc[0][r], sc[1][r]), fmaxf(sc[2][r], sc[3][r]));
#pragma unroll
            for (int mk = 1; mk < 16; mk <<= 1)
                mx = fmaxf(mx, __shfl_xor(mx, mk, 32));
            float nm   = fmaxf(m_i[r], mx);
            float corr = exp2f(m_i[r] - nm);
            float sum  = 0.0f;
#pragma unroll
            for (int nb = 0; nb < 4; ++nb) {
                float e = exp2f(sc[nb][r] - nm);
                sum += e;
                Pl[wv][hi * 8 + r][nb * 16 + lo] = f32_to_f16_bits(e);
            }
#pragma unroll
            for (int mk = 1; mk < 16; mk <<= 1)
                sum += __shfl_xor(sum, mk, 32);
            l_i[r] = l_i[r] * corr + sum;
            m_i[r] = nm;
#pragma unroll
            for (int c = 0; c < 8; ++c) oacc[c][r] *= corr;
        }
        __syncthreads();

        // Two P A-fragments (K ranges 0-31, 32-63)
        Frag pf[2];
#pragma unroll
        for (int hh = 0; hh < 2; ++hh) {
            pf[hh].u4[0] = *(const uint4*)&Pl[wv][lo][hh * 32 + hi * 8];
            pf[hh].u4[1] = *(const uint4*)&Pl[wv][lo][hh * 32 + 16 + hi * 8];
        }

        // O += P @ V : 8 dim-chunks x 2 K-halves -> 16 WMMAs, V frags
        // double-buffered from LDS.
        {
            Frag vfA, vfB;
            loadVfrag(vfA, 0, 0);
#pragma unroll
            for (int i = 0; i < 16; ++i) {
                int c = i >> 1, hh = i & 1;
                if ((i & 1) == 0) {
                    if (i + 1 < 16) loadVfrag(vfB, (i + 1) >> 1, (i + 1) & 1);
                    oacc[c] = __builtin_amdgcn_wmma_f32_16x16x32_f16(
                        false, pf[hh].v, false, vfA.v, (short)0, oacc[c], false, false);
                } else {
                    if (i + 1 < 16) loadVfrag(vfA, (i + 1) >> 1, (i + 1) & 1);
                    oacc[c] = __builtin_amdgcn_wmma_f32_16x16x32_f16(
                        false, pf[hh].v, false, vfB.v, (short)0, oacc[c], false, false);
                }
            }
        }
        __syncthreads();  // WAR: Ks/Vt/Pl reused next iteration
    }

    // Normalize (one reciprocal per row) and write f16 output
#pragma unroll
    for (int r = 0; r < 8; ++r) {
        float inv = 1.0f / l_i[r];
        int t = q0 + wv * 16 + hi * 8 + r;
        unsigned short* orow = O + ((size_t)(b * T + t)) * 2048 + h * 128;
#pragma unroll
        for (int c = 0; c < 8; ++c)
            orow[c * 16 + lo] = f32_to_f16_bits(oacc[c][r] * inv);
    }
}

// ---------------------------------------------------------------------------
// Launcher
// ---------------------------------------------------------------------------
extern "C" void kernel_launch(void* const* d_in, const int* in_sizes, int n_in,
                              void* d_out, int out_size, void* d_ws, size_t ws_size,
                              hipStream_t stream)
{
    (void)in_sizes; (void)n_in; (void)out_size; (void)ws_size;

    const float* x  = (const float*)d_in[0];
    const float* Wq = (const float*)d_in[1];
    const float* bq = (const float*)d_in[2];
    const float* Wk = (const float*)d_in[3];
    const float* bk = (const float*)d_in[4];
    const float* Wv = (const float*)d_in[5];
    const float* bv = (const float*)d_in[6];
    const float* Wo = (const float*)d_in[7];
    const float* bo = (const float*)d_in[8];
    float* out = (float*)d_out;

    const int M = 4096;   // B*T
    const int D = 2048;
    const int DKV = 512;

    char* ws = (char*)d_ws;
    unsigned short* xb  = (unsigned short*)(ws);                 // 16 MB
    unsigned short* Wqb = (unsigned short*)(ws + 16777216);      //  8 MB
    unsigned short* Wkb = (unsigned short*)(ws + 25165824);      //  2 MB
    unsigned short* Wvb = (unsigned short*)(ws + 27262976);      //  2 MB
    unsigned short* Wob = (unsigned short*)(ws + 29360128);      //  8 MB
    unsigned short* Qb  = (unsigned short*)(ws + 37748736);      // 16 MB
    unsigned short* Kb  = (unsigned short*)(ws + 54525952);      //  4 MB
    unsigned short* Vb  = (unsigned short*)(ws + 58720256);      //  4 MB
    unsigned short* AOb = (unsigned short*)(ws + 62914560);      // 16 MB

    // fp32 -> fp16 conversions
    cvt_f32_to_f16<<<(M * D + 255) / 256, 256, 0, stream>>>(x, xb, M * D);
    cvt_f32_to_f16<<<(D * D + 255) / 256, 256, 0, stream>>>(Wq, Wqb, D * D);
    cvt_f32_to_f16<<<(D * DKV + 255) / 256, 256, 0, stream>>>(Wk, Wkb, D * DKV);
    cvt_f32_to_f16<<<(D * DKV + 255) / 256, 256, 0, stream>>>(Wv, Wvb, D * DKV);
    cvt_f32_to_f16<<<(D * D + 255) / 256, 256, 0, stream>>>(Wo, Wob, D * D);

    // Q/K/V projections (128x64 workgroup tiles)
    gemm_f16_wmma<false><<<dim3(D / 64, M / 128), 128, 0, stream>>>(
        xb, Wqb, bq, (void*)Qb, M, D, D);
    gemm_f16_wmma<false><<<dim3(DKV / 64, M / 128), 128, 0, stream>>>(
        xb, Wkb, bk, (void*)Kb, M, DKV, D);
    gemm_f16_wmma<false><<<dim3(DKV / 64, M / 128), 128, 0, stream>>>(
        xb, Wvb, bv, (void*)Vb, M, DKV, D);

    // Attention: (T/64 query blocks, 16 heads, B=2)
    attn_wmma<<<dim3(32, 16, 2), 128, 0, stream>>>(Kb, Qb, Vb, AOb);

    // Output projection -> fp32 out
    gemm_f16_wmma<true><<<dim3(D / 64, M / 128), 128, 0, stream>>>(
        AOb, Wob, bo, (void*)out, M, D, D);
}
